// AdaptiveLowPassLayer_65163243815194
// MI455X (gfx1250) — matrix operand order repaired and checked by reading, not compiled, not run on hardware
//
#include <hip/hip_runtime.h>
#include <hip/hip_bf16.h>
#include <math.h>

// ---------------------------------------------------------------------------
// CDNA5 (gfx1250) WMMA types
// ---------------------------------------------------------------------------
typedef __attribute__((ext_vector_type(16))) __bf16          v16bf;
typedef __attribute__((ext_vector_type(16))) unsigned short  v16us;
typedef __attribute__((ext_vector_type(8)))  float           v8f;

#define BN_EPS 1e-5f
#define PI_F   3.14159265358979f

static __device__ __forceinline__ unsigned short f2bf(float f) {
  // round-to-nearest-even fp32 -> bf16
  unsigned u = __float_as_uint(f);
  u += 0x7FFFu + ((u >> 16) & 1u);
  return (unsigned short)(u >> 16);
}

union FragU { v16us u; v16bf b; };

// ---------------------------------------------------------------------------
// Kernel 1: k=3 conv1d as implicit GEMM via v_wmma_f32_16x16x32_bf16,
// templated on (CIN, COUT) so tile shapes, trip counts and LDS strides are
// compile-time (no runtime idiv, fully unrolled K-chunk loop -> one static
// wmma per chunk, fragment element pairs merge into 32-bit DS accesses).
//   fuse_sc=0 : out = relu(bn(conv(in) + bias))                     (conv1)
//   fuse_sc=1 : out = relu(bn(conv(in,stride) + bias) + shortcut)   (conv2)
//               shortcut = 1x1 conv of sc_in (has_xt) or identity.
// GEMM: M = COUT (pad16), K = CIN*3 (pad32), N = 16 time positions.
// One workgroup = (batch b, 16-wide time tile); wave w owns M-rows [16w,16w+16).
// ---------------------------------------------------------------------------
template<int CIN, int COUT>
__global__ __launch_bounds__(128)
void resconv_wmma(const float* __restrict__ in, const float* __restrict__ sc_in,
                  const float* __restrict__ w,  const float* __restrict__ bias,
                  const float* __restrict__ gamma, const float* __restrict__ beta,
                  const float* __restrict__ mean,  const float* __restrict__ var,
                  const float* __restrict__ xt_w,  const float* __restrict__ xt_b,
                  float* __restrict__ out,
                  int Tin, int Tout, int stride,
                  int fuse_sc, int has_xt, int cin_sc)
{
  constexpr int K      = CIN * 3;
  constexpr int KPAD   = (K + 31) & ~31;
  constexpr int MPAD   = (COUT + 15) & ~15;
  constexpr int NCHUNK = KPAD / 32;

  const int b   = blockIdx.y;
  const int t0  = blockIdx.x * 16;
  const int tid = threadIdx.x;

  __shared__ unsigned short Wl[MPAD * KPAD];   // weights [MPAD][KPAD] bf16
  __shared__ unsigned short Pl[16 * KPAD];     // im2col  [16 n][KPAD k] bf16

  // Stage weights (zero padded) into LDS as bf16.
  for (int e = tid; e < MPAD * KPAD; e += blockDim.x) {
    int m = e / KPAD, k = e - m * KPAD;        // KPAD is a constant
    float v = (m < COUT && k < K) ? w[m * K + k] : 0.f;
    Wl[m * KPAD + k] = f2bf(v);
  }
  // Stage im2col patch (n-major): P[n][k=ci*3+dk] = in[b][ci][(t0+n)*s+dk-1].
  for (int e = tid; e < 16 * KPAD; e += blockDim.x) {
    int n = e / KPAD, k = e - n * KPAD;
    float v = 0.f;
    if (k < K) {
      int ci = k / 3, dk = k - ci * 3;
      int ti = (t0 + n) * stride + dk - 1;
      if (ti >= 0 && ti < Tin) v = in[((size_t)b * CIN + ci) * Tin + ti];
    }
    Pl[n * KPAD + k] = f2bf(v);
  }
  __syncthreads();

  const int wave = tid >> 5;
  const int lane = tid & 31;
  const int row  = lane & 15;     // A: M row within tile / B,C,D: N column
  const int g    = lane >> 4;     // lane half-group

  v8f acc = {};
  const int arow = (wave * 16 + row) * KPAD;
  const int brow = row * KPAD;
#pragma unroll
  for (int kc = 0; kc < NCHUNK; ++kc) {
    const int kb = kc * 32;
    FragU A, Bm;
#pragma unroll
    for (int v = 0; v < 8; ++v) {
      // 16-bit A-matrix 16x32 layout (ISA 7.12.2)
      int kk = (v < 4) ? (2 * v + 8 * g) : (2 * (v - 4) + 8 * g + 16);
      A.u[2 * v]     = Wl[arow + kb + kk];
      A.u[2 * v + 1] = Wl[arow + kb + kk + 1];
      // B-matrix 32x16: same K order as A
      int k2 = kb + kk;
      Bm.u[2 * v]     = Pl[brow + k2];
      Bm.u[2 * v + 1] = Pl[brow + k2 + 1];
    }
    acc = __builtin_amdgcn_wmma_f32_16x16x32_bf16(
        false, A.b, false, Bm.b, (short)0, acc, false, false);
  }

  // Epilogue: C/D layout -> (m = wave*16 + r + 8g, n = lane&15).
#pragma unroll
  for (int r = 0; r < 8; ++r) {
    int m = wave * 16 + r + 8 * g;
    int n = lane & 15;
    int t = t0 + n;
    if (m < COUT && t < Tout) {
      float val = acc[r] + bias[m];
      val = gamma[m] * (val - mean[m]) * rsqrtf(var[m] + BN_EPS) + beta[m];
      if (fuse_sc) {
        float sc;
        if (has_xt) {                       // 1x1 conv shortcut, input length Tin
          sc = xt_b[m];
          for (int ci = 0; ci < cin_sc; ++ci)
            sc += xt_w[m * cin_sc + ci] *
                  sc_in[((size_t)b * cin_sc + ci) * Tin + t * stride];
        } else {                            // identity (stride==1, Tin==Tout)
          sc = sc_in[((size_t)b * cin_sc + m) * Tin + t];
        }
        val += sc;
      }
      val = fmaxf(val, 0.f);
      out[((size_t)b * COUT + m) * Tout + t] = val;
    }
  }
}

// ---------------------------------------------------------------------------
// Kernel 2: FC1 (2048 -> 512) + BN + ReLU as bf16 WMMA GEMM.
//   out[b][o] = relu(bn(sum_k h[b][k] * w[o][k] + bias[o]))
// Tile: 64 output features (4 waves) x 16 batch rows; K = 2048 in 64 chunks.
// ---------------------------------------------------------------------------
__global__ __launch_bounds__(128)
void fc1_wmma(const float* __restrict__ h, const float* __restrict__ w,
              const float* __restrict__ bias,
              const float* __restrict__ gamma, const float* __restrict__ beta,
              const float* __restrict__ mean,  const float* __restrict__ var,
              float* __restrict__ out)
{
  const int o0  = blockIdx.x * 64;
  const int b0  = blockIdx.y * 16;
  const int tid = threadIdx.x;
  __shared__ unsigned short Wt[64 * 32];   // [m][k]
  __shared__ unsigned short Ht[16 * 32];   // [n][k] (n-major: K pairs contiguous)
  const int wave = tid >> 5, lane = tid & 31, g = lane >> 4, row = lane & 15;

  v8f acc = {};
  for (int kc = 0; kc < 64; ++kc) {
    const int kb = kc * 32;
    for (int e = tid; e < 64 * 32; e += 128) {
      int m = e >> 5, k = e & 31;
      Wt[e] = f2bf(w[(size_t)(o0 + m) * 2048 + kb + k]);
    }
    for (int e = tid; e < 16 * 32; e += 128) {
      int n = e >> 5, k = e & 31;
      Ht[e] = f2bf(h[(size_t)(b0 + n) * 2048 + kb + k]);
    }
    __syncthreads();
    FragU A, Bm;
#pragma unroll
    for (int v = 0; v < 8; ++v) {
      int kk = (v < 4) ? (2 * v + 8 * g) : (2 * (v - 4) + 8 * g + 16);
      A.u[2 * v]     = Wt[(wave * 16 + row) * 32 + kk];
      A.u[2 * v + 1] = Wt[(wave * 16 + row) * 32 + kk + 1];
      int k2 = kk;
      Bm.u[2 * v]     = Ht[row * 32 + k2];
      Bm.u[2 * v + 1] = Ht[row * 32 + k2 + 1];
    }
    acc = __builtin_amdgcn_wmma_f32_16x16x32_bf16(
        false, A.b, false, Bm.b, (short)0, acc, false, false);
    __syncthreads();
  }
#pragma unroll
  for (int r = 0; r < 8; ++r) {
    int m  = o0 + wave * 16 + r + 8 * g;       // output feature (global)
    int bb = b0 + (lane & 15);                 // batch row
    float val = acc[r] + bias[m];
    val = gamma[m] * (val - mean[m]) * rsqrtf(var[m] + BN_EPS) + beta[m];
    val = fmaxf(val, 0.f);
    out[(size_t)bb * 512 + m] = val;
  }
}

// ---------------------------------------------------------------------------
// Kernel 3: FC2 + sigmoid -> cutoff, then build normalized windowed-sinc taps.
// One workgroup per batch sample.
// ---------------------------------------------------------------------------
__global__ __launch_bounds__(128)
void fc2_build_kernel(const float* __restrict__ h1, const float* __restrict__ w2,
                      const float* __restrict__ b2, float* __restrict__ kern)
{
  const int b   = blockIdx.x;
  const int tid = threadIdx.x;
  __shared__ float red[128];
  __shared__ float taps[101];
  __shared__ float fc_s;

  float p = 0.f;
  for (int j = tid; j < 512; j += 128) p += h1[(size_t)b * 512 + j] * w2[j];
  red[tid] = p;
  __syncthreads();
  for (int s = 64; s > 0; s >>= 1) {
    if (tid < s) red[tid] += red[tid + s];
    __syncthreads();
  }
  if (tid == 0) {
    float z   = red[0] + b2[0];
    float sig = 1.f / (1.f + expf(-z));
    fc_s = (300.f + 250.f * sig) / 2048.f;    // fc_hz / FS
  }
  __syncthreads();
  float fc = fc_s;
  if (tid < 101) {
    float t   = (float)(tid - 50);
    float arg = 2.f * fc * t;
    float snc = (arg == 0.f) ? 1.f : sinf(PI_F * arg) / (PI_F * arg);
    float win = 0.54f - 0.46f * cosf(2.f * PI_F * (float)tid / 101.f);
    taps[tid] = 2.f * fc * snc * win;
  }
  __syncthreads();
  red[tid] = (tid < 101) ? taps[tid] : 0.f;
  __syncthreads();
  for (int s = 64; s > 0; s >>= 1) {
    if (tid < s) red[tid] += red[tid + s];
    __syncthreads();
  }
  float inv = 1.f / red[0];
  if (tid < 101) kern[(size_t)b * 101 + tid] = taps[tid] * inv;
}

// ---------------------------------------------------------------------------
// Kernel 4: per-sample 101-tap depthwise FIR ('same' zero padding), LDS tiled.
// One workgroup per (b, c) row of 2048 samples.
// ---------------------------------------------------------------------------
__global__ __launch_bounds__(256)
void fir_apply(const float* __restrict__ x, const float* __restrict__ kern,
               float* __restrict__ y)
{
  const int row = blockIdx.x;      // b*2 + c
  const int b   = row >> 1;
  const int tid = threadIdx.x;
  __shared__ float xs[2048 + 100];
  __shared__ float kf[101];
  if (tid < 101) kf[tid] = kern[(size_t)b * 101 + tid];
  for (int i = tid; i < 2148; i += 256) {
    int gp = i - 50;
    xs[i] = (gp >= 0 && gp < 2048) ? x[(size_t)row * 2048 + gp] : 0.f;
  }
  __syncthreads();
  for (int t = tid; t < 2048; t += 256) {
    float s = 0.f;
#pragma unroll 4
    for (int j = 0; j < 101; ++j) s += kf[j] * xs[t + j];
    y[(size_t)row * 2048 + t] = s;
  }
}

// ---------------------------------------------------------------------------
// Host orchestration
// ---------------------------------------------------------------------------
struct ConvArgs {
  const float *in, *sc_in, *w, *bias, *gamma, *beta, *mean, *var, *xw, *xb;
  float* out;
  int Tin, Tout, stride, fuse_sc, has_xt, cin_sc;
};

template<int CIN, int COUT>
static void launch_conv(const ConvArgs& a, hipStream_t stream) {
  constexpr int WAVES = ((COUT + 15) / 16);
  resconv_wmma<CIN, COUT><<<dim3(a.Tout / 16, 1024), dim3(32 * WAVES), 0, stream>>>(
      a.in, a.sc_in, a.w, a.bias, a.gamma, a.beta, a.mean, a.var, a.xw, a.xb,
      a.out, a.Tin, a.Tout, a.stride, a.fuse_sc, a.has_xt, a.cin_sc);
}

static void conv_dispatch(int cin, int cout, const ConvArgs& a, hipStream_t s) {
  if      (cin ==  2 && cout ==  8) launch_conv< 2,  8>(a, s);
  else if (cin ==  8 && cout ==  8) launch_conv< 8,  8>(a, s);
  else if (cin ==  8 && cout == 16) launch_conv< 8, 16>(a, s);
  else if (cin == 16 && cout == 16) launch_conv<16, 16>(a, s);
  else if (cin == 16 && cout == 32) launch_conv<16, 32>(a, s);
  else if (cin == 32 && cout == 32) launch_conv<32, 32>(a, s);
  else if (cin == 32 && cout == 64) launch_conv<32, 64>(a, s);
  else if (cin == 64 && cout == 64) launch_conv<64, 64>(a, s);
}

extern "C" void kernel_launch(void* const* d_in, const int* in_sizes, int n_in,
                              void* d_out, int out_size, void* d_ws, size_t ws_size,
                              hipStream_t stream)
{
  (void)in_sizes; (void)n_in; (void)out_size; (void)ws_size;

  static const int CIN[16]  = {2,8,8,8,16,16,16,16,32,32,32,32,32,32,64,64};
  static const int COUT[16] = {8,8,8,16,16,16,16,32,32,32,32,32,32,64,64,64};
  static const int STR[16]  = {1,1,1,4,1,1,1,4,1,1,1,1,1,4,1,1};
  static const int TIN[16]  = {2048,2048,2048,2048,512,512,512,512,
                               128,128,128,128,128,128,32,32};

  const float* x = (const float*)d_in[0];
  int p = 1;
  const float *w1[16], *b1[16], *g1[16], *be1[16], *m1[16], *v1[16];
  const float *w2[16], *b2[16], *g2[16], *be2[16], *m2[16], *v2[16];
  const float *xw[16], *xb[16];
  for (int i = 0; i < 16; ++i) {
    w1[i] = (const float*)d_in[p++]; b1[i] = (const float*)d_in[p++];
    g1[i] = (const float*)d_in[p++]; be1[i] = (const float*)d_in[p++];
    m1[i] = (const float*)d_in[p++]; v1[i] = (const float*)d_in[p++];
    w2[i] = (const float*)d_in[p++]; b2[i] = (const float*)d_in[p++];
    g2[i] = (const float*)d_in[p++]; be2[i] = (const float*)d_in[p++];
    m2[i] = (const float*)d_in[p++]; v2[i] = (const float*)d_in[p++];
    bool hx = (CIN[i] != COUT[i]) || (STR[i] > 1);
    if (hx) { xw[i] = (const float*)d_in[p++]; xb[i] = (const float*)d_in[p++]; }
    else    { xw[i] = nullptr;                 xb[i] = nullptr; }
  }
  const float* fc1w = (const float*)d_in[p++];
  const float* fc1b = (const float*)d_in[p++];
  const float* gfc  = (const float*)d_in[p++];
  const float* befc = (const float*)d_in[p++];
  const float* mfc  = (const float*)d_in[p++];
  const float* vfc  = (const float*)d_in[p++];
  const float* fc2w = (const float*)d_in[p++];
  const float* fc2b = (const float*)d_in[p++];

  // Workspace layout (fp32): two activation ping-pong buffers, conv1 tmp,
  // fc1 output, per-sample FIR taps.
  float* bufA = (float*)d_ws;
  float* bufB = bufA + (size_t)1024 * 8 * 2048;     // 64 MB
  float* bufT = bufB + (size_t)1024 * 8 * 2048;     // 64 MB
  float* fc1o = bufT + (size_t)1024 * 16 * 2048;    // 128 MB tmp (max conv1 out)
  float* kern = fc1o + (size_t)1024 * 512;

  const float* cur = x;
  float* pong[2] = {bufA, bufB};
  int which = 0;
  for (int i = 0; i < 16; ++i) {
    const int Tin = TIN[i], s = STR[i], Tout = Tin / s;
    const int cin = CIN[i], cout = COUT[i];
    // conv1 (stride 1) + bn1 + relu -> bufT  [1024, cout, Tin]
    ConvArgs a1 = { cur, nullptr, w1[i], b1[i], g1[i], be1[i], m1[i], v1[i],
                    nullptr, nullptr, bufT, Tin, Tin, 1, 0, 0, 0 };
    conv_dispatch(cin, cout, a1, stream);
    // conv2 (stride s) + bn2 + shortcut + relu -> next activation
    float* nxt = pong[which]; which ^= 1;
    const int hx = (xw[i] != nullptr) ? 1 : 0;
    ConvArgs a2 = { bufT, cur, w2[i], b2[i], g2[i], be2[i], m2[i], v2[i],
                    xw[i], xb[i], nxt, Tin, Tout, s, 1, hx, cin };
    conv_dispatch(cout, cout, a2, stream);
    cur = nxt;
  }

  // FC1 + BN + ReLU : [1024,2048] x [512,2048]^T -> [1024,512]
  fc1_wmma<<<dim3(8, 64), dim3(128), 0, stream>>>(
      cur, fc1w, fc1b, gfc, befc, mfc, vfc, fc1o);

  // FC2 + sigmoid + normalized windowed-sinc taps per sample
  fc2_build_kernel<<<dim3(1024), dim3(128), 0, stream>>>(fc1o, fc2w, fc2b, kern);

  // Depthwise 101-tap FIR over original input -> d_out
  fir_apply<<<dim3(2048), dim3(256), 0, stream>>>(x, kern, (float*)d_out);
}
